// MultiheadAttention_66735201845498
// MI455X (gfx1250) — matrix operand order repaired and checked by reading, not compiled, not run on hardware
//
#include <hip/hip_runtime.h>
#include <hip/hip_bf16.h>
#include <math.h>

typedef __attribute__((ext_vector_type(16))) _Float16 v16h_t;
typedef __attribute__((ext_vector_type(8)))  _Float16 v8h_t;
typedef __attribute__((ext_vector_type(8)))  float    v8f_t;
typedef __attribute__((ext_vector_type(4)))  float    f32x4_t;
typedef unsigned int u32x4_t __attribute__((ext_vector_type(4)));
typedef int          i32x4_t __attribute__((ext_vector_type(4)));
typedef int          i32x8_t __attribute__((ext_vector_type(8)));

#define B_  4
#define L_  2048
#define E_  512
#define H_  8
#define DH_ 64

// ---------------------------------------------------------------------------
// Phase 1: Q/K/V projection.  out = x @ W^T + b  on v_wmma_f32_16x16x32_f16.
// Each wave owns a 16(M) x 64(N) tile.  z = blockIdx.z selects {Q, K, V}.
// Q,K stored [B,H,L,DH] f16;  V stored transposed [B,H,DH,L] f16 so the
// phase-2 P@V B-fragments are contiguous.
// ---------------------------------------------------------------------------
__global__ __launch_bounds__(256) void mha_qkv_proj_kernel(
    const float* __restrict__ xq, const float* __restrict__ xkv,
    const float* __restrict__ Wq, const float* __restrict__ bq,
    const float* __restrict__ Wk, const float* __restrict__ bk,
    const float* __restrict__ Wv, const float* __restrict__ bv,
    _Float16* __restrict__ Qh, _Float16* __restrict__ Kh,
    _Float16* __restrict__ Vt)
{
  const int lane = threadIdx.x & 31;
  const int wave = threadIdx.x >> 5;
  const int ln   = lane & 15;       // column / row-in-tile selector
  const int g    = lane >> 4;       // lane half

  const int z = blockIdx.z;
  const float* x    = (z == 0) ? xq : xkv;
  const float* W    = (z == 0) ? Wq : (z == 1) ? Wk : Wv;
  const float* bias = (z == 0) ? bq : (z == 1) ? bk : bv;

  const int m0 = (blockIdx.x * 8 + wave) * 16;   // row tile in [0, B*L)
  const int n0 = blockIdx.y * 64;                // col tile in [0, E)

  v8f_t acc[4];
#pragma unroll
  for (int nt = 0; nt < 4; ++nt) {
    float bb = bias[n0 + nt * 16 + ln];
#pragma unroll
    for (int r = 0; r < 8; ++r) acc[nt][r] = bb;
  }

  const float* arow = x + (size_t)(m0 + ln) * E_;
  for (int k0 = 0; k0 < E_; k0 += 32) {
    // ---- A fragment (16x32 f16): lane ln = row, chunks K=[8g,8g+8) and [16+8g,..)
    v16h_t a;
    {
      const f32x4_t* p0 = (const f32x4_t*)(arow + k0 + g * 8);
      const f32x4_t* p1 = (const f32x4_t*)(arow + k0 + 16 + g * 8);
      f32x4_t c0 = p0[0], c1 = p0[1], c2 = p1[0], c3 = p1[1];
#pragma unroll
      for (int i = 0; i < 4; ++i) {
        a[i]      = (_Float16)c0[i];
        a[4 + i]  = (_Float16)c1[i];
        a[8 + i]  = (_Float16)c2[i];
        a[12 + i] = (_Float16)c3[i];
      }
    }
#pragma unroll
    for (int nt = 0; nt < 4; ++nt) {
      // ---- B fragment (32x16 f16): column N = row n of W (x@W^T), contiguous
      const float* wrow = W + (size_t)(n0 + nt * 16 + ln) * E_ + k0;
      const f32x4_t* wp = (const f32x4_t*)(wrow + g * 8);
      const f32x4_t* wq = (const f32x4_t*)(wrow + 16 + g * 8);
      f32x4_t w0 = wp[0], w1 = wp[1], w2 = wq[0], w3 = wq[1];
      v16h_t bf;
#pragma unroll
      for (int i = 0; i < 4; ++i) {
        bf[i]      = (_Float16)w0[i];
        bf[4 + i]  = (_Float16)w1[i];
        bf[8 + i]  = (_Float16)w2[i];
        bf[12 + i] = (_Float16)w3[i];
      }
      acc[nt] = __builtin_amdgcn_wmma_f32_16x16x32_f16(
          false, a, false, bf, (short)0, acc[nt], false, false);
    }
  }

  // ---- store: C layout -> VGPR r holds M = r + 8*g, N = ln
#pragma unroll
  for (int nt = 0; nt < 4; ++nt) {
#pragma unroll
    for (int r = 0; r < 8; ++r) {
      int m = m0 + r + 8 * g;
      int n = n0 + nt * 16 + ln;
      int bidx = m >> 11;          // / L_
      int lidx = m & (L_ - 1);
      int h = n >> 6;              // / DH_
      int d = n & 63;
      _Float16 val = (_Float16)acc[nt][r];
      if (z == 0)
        Qh[(((size_t)bidx * H_ + h) * L_ + lidx) * DH_ + d] = val;
      else if (z == 1)
        Kh[(((size_t)bidx * H_ + h) * L_ + lidx) * DH_ + d] = val;
      else
        Vt[(((size_t)bidx * H_ + h) * DH_ + d) * L_ + lidx] = val;
    }
  }
}

// ---------------------------------------------------------------------------
// TDM helper: async-load a 64x64 f16 tile (row stride `stride_elems`) from
// global memory into LDS at byte offset `lds_off`.  D# built per ISA 8.3/8.4:
// group0 = {count=1 | flags, lds_addr, global_addr[31:0],
//           global_addr[56:32] | type=2<<30}
// group1 = {data_size=2B, tensor_dim0=64, tensor_dim1=64, tile 64x64,
//           tensor_dim0_stride = stride_elems}
// ---------------------------------------------------------------------------
__device__ inline void tdm_load_tile64x64(unsigned int lds_off,
                                          const _Float16* gsrc,
                                          unsigned int stride_elems) {
  unsigned long long ga = (unsigned long long)(uintptr_t)gsrc;
  u32x4_t g0;
  g0[0] = 1u;                                            // count=1, no gather
  g0[1] = lds_off;                                       // lds_addr (bytes)
  g0[2] = (unsigned int)ga;                              // global_addr[31:0]
  g0[3] = (unsigned int)((ga >> 32) & 0x01FFFFFFull)     // global_addr[56:32]
          | (2u << 30);                                  // type = 2 ("image")
  i32x8_t g1;
  g1[0] = (int)(1u << 16);        // workgroup_mask=0, data_size=1 (2 bytes)
  g1[1] = (int)(64u << 16);       // tensor_dim0[15:0] = 64   (bits 79:48)
  g1[2] = (int)(64u << 16);       // tensor_dim1[15:0] = 64   (bits 111:80)
  g1[3] = (int)(64u << 16);       // tile_dim0 = 64           (bits 127:112)
  g1[4] = 64;                     // tile_dim1 = 64, tile_dim2 = 0
  g1[5] = (int)stride_elems;      // tensor_dim0_stride[31:0]
  g1[6] = 0;                      // stride0[47:32]=0, tensor_dim1_stride lo=0
  g1[7] = 0;
  i32x4_t z4 = {0, 0, 0, 0};
#if defined(__clang_major__) && (__clang_major__ >= 23)
  i32x8_t z8 = {0, 0, 0, 0, 0, 0, 0, 0};
  __builtin_amdgcn_tensor_load_to_lds(g0, g1, z4, z4, z8, 0);
#else
  __builtin_amdgcn_tensor_load_to_lds(g0, g1, z4, z4, 0);
#endif
}

// ---------------------------------------------------------------------------
// Phase 2: flash attention.  All 8 waves of a block share one (b,h) and walk
// the same 64-key blocks, so K/V tiles are staged once per block into LDS by
// the Tensor Data Mover (wave 0 issues tensor_load_to_lds, waits TENSORcnt,
// block barrier).  S = Q K^T and O += P V on v_wmma_f32_16x16x32_f16 with
// B-fragments read from the LDS tiles; P bounced C-layout -> A-layout through
// a wave-private LDS region.
// ---------------------------------------------------------------------------
__global__ __launch_bounds__(256) void mha_flash_attn_kernel(
    const _Float16* __restrict__ Qh, const _Float16* __restrict__ Kh,
    const _Float16* __restrict__ Vt, const unsigned char* __restrict__ mask,
    float* __restrict__ out)
{
  __shared__ _Float16 ldsK[64 * 64];       // 8 KB: K block, row-major [k][d]
  __shared__ _Float16 ldsV[64 * 64];       // 8 KB: V^T block, row-major [d][k]
  __shared__ _Float16 lds_p[8][16 * 72];   // per-wave P bounce (72 = 64+8 pad)

  const int lane = threadIdx.x & 31;
  const int wave = threadIdx.x >> 5;
  const int ln   = lane & 15;
  const int g    = lane >> 4;

  const int h  = blockIdx.y;
  const int b  = blockIdx.z;
  const int q0 = (blockIdx.x * 8 + wave) * 16;
  const size_t bh = (size_t)b * H_ + h;
  const _Float16* Qbh = Qh + bh * (size_t)L_ * DH_;
  const _Float16* Kbh = Kh + bh * (size_t)L_ * DH_;
  const _Float16* Vbh = Vt + bh * (size_t)DH_ * L_;
  const unsigned char* mrow = mask + (size_t)b * L_;

  const unsigned int ldsK_off = (unsigned int)(uintptr_t)(&ldsK[0]);
  const unsigned int ldsV_off = (unsigned int)(uintptr_t)(&ldsV[0]);

  // ---- load Q A-fragments for the two K-steps of DH=64 (kept in VGPRs)
  v16h_t qa[2];
  {
    const _Float16* qrow = Qbh + (size_t)(q0 + ln) * DH_;
#pragma unroll
    for (int ks = 0; ks < 2; ++ks) {
      v8h_t c0 = *(const v8h_t*)(qrow + ks * 32 + g * 16);
      v8h_t c1 = *(const v8h_t*)(qrow + ks * 32 + g * 16 + 8);
#pragma unroll
      for (int i = 0; i < 8; ++i) { qa[ks][i] = c0[i]; qa[ks][8 + i] = c1[i]; }
    }
  }

  float mx[8], lsum[8];
#pragma unroll
  for (int r = 0; r < 8; ++r) { mx[r] = -1e30f; lsum[r] = 0.f; }
  v8f_t o[4];
#pragma unroll
  for (int dt = 0; dt < 4; ++dt)
#pragma unroll
    for (int r = 0; r < 8; ++r) o[dt][r] = 0.f;

  const float scale = 0.125f;  // 1/sqrt(DH)

  for (int kb = 0; kb < L_; kb += 64) {
    // ---- TDM: stage this key block's K (64x64) and V^T (64x64) into LDS
    if (wave == 0) {
      tdm_load_tile64x64(ldsK_off, Kbh + (size_t)kb * DH_, DH_);  // rows = keys
      tdm_load_tile64x64(ldsV_off, Vbh + kb, L_);                 // rows = d
      __builtin_amdgcn_s_wait_tensorcnt(0);
    }
    __syncthreads();   // tiles visible to all waves

    // ---- S = Q K^T over this 16x64 key block (B column n = key row in LDS)
    v8f_t s[4];
#pragma unroll
    for (int nt = 0; nt < 4; ++nt) {
      v8f_t acc;
#pragma unroll
      for (int r = 0; r < 8; ++r) acc[r] = 0.f;
#pragma unroll
      for (int ks = 0; ks < 2; ++ks) {
        v16h_t kf = *(const v16h_t*)(&ldsK[(nt * 16 + ln) * 64 + ks * 32 + g * 16]);
        acc = __builtin_amdgcn_wmma_f32_16x16x32_f16(
            false, qa[ks], false, kf, (short)0, acc, false, false);
      }
      int masked = mrow[kb + nt * 16 + ln];   // key_padding_mask, per key column
#pragma unroll
      for (int r = 0; r < 8; ++r)
        s[nt][r] = masked ? -1e30f : acc[r] * scale;
    }

    // ---- online softmax (row = M lives in one 16-lane half -> xor 1/2/4/8)
    float mnew[8], esc[8];
#pragma unroll
    for (int r = 0; r < 8; ++r) {
      float rm = fmaxf(fmaxf(s[0][r], s[1][r]), fmaxf(s[2][r], s[3][r]));
      rm = fmaxf(rm, __shfl_xor(rm, 1, 32));
      rm = fmaxf(rm, __shfl_xor(rm, 2, 32));
      rm = fmaxf(rm, __shfl_xor(rm, 4, 32));
      rm = fmaxf(rm, __shfl_xor(rm, 8, 32));
      mnew[r] = fmaxf(mx[r], rm);
      esc[r]  = __expf(mx[r] - mnew[r]);
      mx[r]   = mnew[r];
    }
#pragma unroll
    for (int r = 0; r < 8; ++r) {
      float rs = 0.f;
#pragma unroll
      for (int nt = 0; nt < 4; ++nt) {
        float pv = __expf(s[nt][r] - mnew[r]);
        s[nt][r] = pv;
        rs += pv;
      }
      rs += __shfl_xor(rs, 1, 32);
      rs += __shfl_xor(rs, 2, 32);
      rs += __shfl_xor(rs, 4, 32);
      rs += __shfl_xor(rs, 8, 32);
      lsum[r] = lsum[r] * esc[r] + rs;
#pragma unroll
      for (int dt = 0; dt < 4; ++dt) o[dt][r] *= esc[r];
    }

    // ---- P: C-layout -> row-major LDS (wave-private region; DS is in-order
    //      per wave, so no barrier needed before the A-fragment reload)
#pragma unroll
    for (int nt = 0; nt < 4; ++nt)
#pragma unroll
      for (int r = 0; r < 8; ++r)
        lds_p[wave][(r + 8 * g) * 72 + nt * 16 + ln] = (_Float16)s[nt][r];

    // ---- O += P V  (B fragment = V^T row d from LDS tile)
#pragma unroll
    for (int ks = 0; ks < 2; ++ks) {
      const _Float16* prow = &lds_p[wave][ln * 72];
      v8h_t c0 = *(const v8h_t*)(prow + ks * 32 + g * 16);
      v8h_t c1 = *(const v8h_t*)(prow + ks * 32 + g * 16 + 8);
      v16h_t pa;
#pragma unroll
      for (int i = 0; i < 8; ++i) { pa[i] = c0[i]; pa[8 + i] = c1[i]; }
#pragma unroll
      for (int dt = 0; dt < 4; ++dt) {
        v16h_t vf = *(const v16h_t*)(&ldsV[(dt * 16 + ln) * 64 + ks * 32 + g * 16]);
        o[dt] = __builtin_amdgcn_wmma_f32_16x16x32_f16(
            false, pa, false, vf, (short)0, o[dt], false, false);
      }
    }
    __syncthreads();   // everyone done with ldsK/ldsV before next TDM overwrite
  }

  // ---- normalize and store ctx [B, L, H*DH] f32
#pragma unroll
  for (int r = 0; r < 8; ++r) {
    float inv = (lsum[r] > 0.f) ? 1.f / lsum[r] : 0.f;
    int q = q0 + r + 8 * g;
    float* orow = out + ((size_t)b * L_ + q) * E_ + h * DH_;
#pragma unroll
    for (int dt = 0; dt < 4; ++dt)
      orow[dt * 16 + ln] = o[dt][r] * inv;
  }
}

// ---------------------------------------------------------------------------
extern "C" void kernel_launch(void* const* d_in, const int* in_sizes, int n_in,
                              void* d_out, int out_size, void* d_ws, size_t ws_size,
                              hipStream_t stream) {
  const float* xq  = (const float*)d_in[0];
  const float* xkv = (const float*)d_in[1];
  const unsigned char* mask = (const unsigned char*)d_in[2];
  const float* Wq = (const float*)d_in[3];
  const float* bq = (const float*)d_in[4];
  const float* Wk = (const float*)d_in[5];
  const float* bk = (const float*)d_in[6];
  const float* Wv = (const float*)d_in[7];
  const float* bv = (const float*)d_in[8];
  float* out = (float*)d_out;

  const size_t per = (size_t)B_ * H_ * L_ * DH_;   // 4 Mi f16 elems each
  _Float16* Qh = (_Float16*)d_ws;
  _Float16* Kh = Qh + per;
  _Float16* Vt = Kh + per;

  dim3 gridP(64, 8, 3);          // 512 M-tiles / 8 waves, 8 N-groups, {Q,K,V}
  mha_qkv_proj_kernel<<<gridP, 256, 0, stream>>>(xq, xkv, Wq, bq, Wk, bk, Wv, bv,
                                                 Qh, Kh, Vt);

  dim3 gridA(L_ / (16 * 8), H_, B_);  // 16 x 8 x 4 blocks, 8 waves each
  mha_flash_attn_kernel<<<gridA, 256, 0, stream>>>(Qh, Kh, Vt, mask, out);
}
